// DAGAttention_2293512536385
// MI455X (gfx1250) — compile-verified, matrix operand order checked by reading, not run on hardware
//
#include <hip/hip_runtime.h>
#include <hip/hip_bf16.h>

// ---------------- problem constants ----------------
#define NROWS 65536
#define EMB   1024
#define K2    2048      // 2*EMB
#define ATT   512
#define KC    32        // K per bf16 WMMA
#define NKC   (K2/KC)   // 64 K-chunks
#define NT    (ATT/16)  // 32 N-tiles of 16 cols
#define NTG   16        // N-tiles per register group (acc = 128 VGPRs)
#define NGRP  (NT/NTG)  // 2 groups

typedef __attribute__((ext_vector_type(16))) __bf16 v16bf;
typedef __attribute__((ext_vector_type(2)))  __bf16 v2bf;
typedef __attribute__((ext_vector_type(8)))  float  v8f;

union Frag16 {
    uint4          q[2];
    v16bf          v;
    v2bf           p[8];
    unsigned short u[16];
};

__device__ __forceinline__ unsigned short f2bf(float f) {
    unsigned int u = __float_as_uint(f);
    unsigned int r = u + 0x7FFFu + ((u >> 16) & 1u);   // round-to-nearest-even
    return (unsigned short)(r >> 16);
}

__device__ __forceinline__ void pack2(Frag16& fr, int i, float lo, float hi) {
#if __has_builtin(__builtin_amdgcn_cvt_pk_bf16_f32)
    fr.p[i] = __builtin_amdgcn_cvt_pk_bf16_f32(lo, hi); // v_cvt_pk_bf16_f32
#else
    fr.u[2 * i]     = f2bf(lo);
    fr.u[2 * i + 1] = f2bf(hi);
#endif
}

// branchless fast tanh: 1 - 2/(2^(2*log2e*x)+1); exact limits at +/-inf
__device__ __forceinline__ float fast_tanh(float x) {
#if __has_builtin(__builtin_amdgcn_exp2f) && __has_builtin(__builtin_amdgcn_rcpf)
    float e = __builtin_amdgcn_exp2f(x * 2.8853900817779268f);
    return 1.0f - 2.0f * __builtin_amdgcn_rcpf(e + 1.0f);
#else
    float e = __expf(2.0f * x);
    return 1.0f - 2.0f / (e + 1.0f);
#endif
}

// ---------------- kernel 1: W1 [K2,ATT] f32 -> W1T [ATT,K2] bf16 ----------------
__global__ __launch_bounds__(256) void convert_w1(const float* __restrict__ w1,
                                                  unsigned short* __restrict__ w1t) {
    int idx = blockIdx.x * 256 + threadIdx.x;          // over K2*ATT
    int k = idx >> 9;                                  // / ATT
    int n = idx & (ATT - 1);
    w1t[(size_t)n * K2 + k] = f2bf(w1[idx]);
}

// ---------------- kernel 2: scores = tanh(x@W1+b1) @ W2 + b2 ----------------
// block = 256 threads (8 waves), each wave owns 16 rows; grid = NROWS/128
__global__ __launch_bounds__(256) void scores_kernel(
    const float* __restrict__ leaves, const float* __restrict__ anc,
    const unsigned short* __restrict__ w1t,
    const float* __restrict__ b1p, const float* __restrict__ b2p,
    const float* __restrict__ w2p, float* __restrict__ scores)
{
    const int lane    = threadIdx.x & 31;
    const int wave    = threadIdx.x >> 5;              // 0..7
    const int rowbase = blockIdx.x * 128 + wave * 16;
    const int half    = lane >> 4;                     // 0/1
    const int l15     = lane & 15;
    const int m       = rowbase + l15;                 // A-fragment row for this lane

    float sp[8];
    #pragma unroll
    for (int r = 0; r < 8; ++r) sp[r] = 0.f;

    for (int g = 0; g < NGRP; ++g) {
        v8f acc[NTG];
        #pragma unroll
        for (int t = 0; t < NTG; ++t) acc[t] = (v8f)0.f;

        for (int kc = 0; kc < NKC; ++kc) {
            // ----- A fragment (ISA 16-bit A 16x32 layout) -----
            const int kb = kc * KC + half * 8;         // K base for this lane
            const float* src = (kb < EMB)
                ? (leaves + (size_t)m * EMB + kb)
                : (anc    + (size_t)m * EMB + (kb - EMB));
            float4 f0 = *(const float4*)(src);
            float4 f1 = *(const float4*)(src + 4);
            float4 f2 = *(const float4*)(src + 16);
            float4 f3 = *(const float4*)(src + 20);
            __builtin_prefetch(src + KC, 0, 1);        // global_prefetch next K-chunk

            Frag16 fa;
            pack2(fa, 0, f0.x, f0.y);
            pack2(fa, 1, f0.z, f0.w);
            pack2(fa, 2, f1.x, f1.y);
            pack2(fa, 3, f1.z, f1.w);
            pack2(fa, 4, f2.x, f2.y);
            pack2(fa, 5, f2.z, f2.w);
            pack2(fa, 6, f3.x, f3.y);
            pack2(fa, 7, f3.z, f3.w);

            #pragma unroll
            for (int t = 0; t < NTG; ++t) {
                // ----- B fragment: column = nt*16 + l15, contiguous along K in W1T -----
                const int nt = g * NTG + t;
                const unsigned short* bp =
                    w1t + (size_t)(nt * 16 + l15) * K2 + kb;
                Frag16 fb;
                fb.q[0] = *(const uint4*)(bp);
                fb.q[1] = *(const uint4*)(bp + 16);
                acc[t] = __builtin_amdgcn_wmma_f32_16x16x32_bf16(
                    false, fa.v, false, fb.v, (short)0, acc[t], false, false);
            }
        }

        // ----- epilogue: sp[r] += tanh(h + b1[col]) * W2[col] -----
        #pragma unroll
        for (int t = 0; t < NTG; ++t) {
            const int col = (g * NTG + t) * 16 + l15;  // C/D layout: lane -> column
            const float b1c = b1p[col];
            const float w2c = w2p[col];
            #pragma unroll
            for (int r = 0; r < 8; ++r)
                sp[r] += fast_tanh(acc[t][r] + b1c) * w2c;
        }
    }

    // reduce over the 16 lanes of each half (rows r / r+8)
    #pragma unroll
    for (int r = 0; r < 8; ++r) {
        float v = sp[r];
        #pragma unroll
        for (int off = 8; off >= 1; off >>= 1)
            v += __shfl_xor(v, off, 32);
        sp[r] = v;
    }
    if (l15 == 0) {
        const float bb = b2p[0];
        #pragma unroll
        for (int r = 0; r < 8; ++r)
            scores[rowbase + r + 8 * half] = sp[r] + bb;
    }
}

// ---------------- kernel 3: softmax stats (max, 1/sum) ----------------
__global__ __launch_bounds__(1024) void softmax_stats(const float* __restrict__ scores,
                                                      float* __restrict__ stats) {
    __shared__ float red[1024];
    float mx = -3.4e38f;
    for (int i = threadIdx.x; i < NROWS; i += 1024) mx = fmaxf(mx, scores[i]);
    red[threadIdx.x] = mx; __syncthreads();
    for (int s = 512; s > 0; s >>= 1) {
        if (threadIdx.x < s) red[threadIdx.x] = fmaxf(red[threadIdx.x], red[threadIdx.x + s]);
        __syncthreads();
    }
    const float gmax = red[0]; __syncthreads();
    float sum = 0.f;
    for (int i = threadIdx.x; i < NROWS; i += 1024) sum += expf(scores[i] - gmax);
    red[threadIdx.x] = sum; __syncthreads();
    for (int s = 512; s > 0; s >>= 1) {
        if (threadIdx.x < s) red[threadIdx.x] += red[threadIdx.x + s];
        __syncthreads();
    }
    if (threadIdx.x == 0) { stats[0] = gmax; stats[1] = 1.0f / red[0]; }
}

// ---------------- kernel 4: attn[i] = exp(s-max)/sum ----------------
__global__ __launch_bounds__(256) void attn_kernel(const float* __restrict__ scores,
                                                   const float* __restrict__ stats,
                                                   float* __restrict__ attn) {
    int i = blockIdx.x * 256 + threadIdx.x;
    attn[i] = expf(scores[i] - stats[0]) * stats[1];
}

// ---------------- kernel 5: out[e] += sum_i attn[i]*anc[i][e] ----------------
__global__ __launch_bounds__(256) void wsum_kernel(const float* __restrict__ anc,
                                                   const float* __restrict__ attn,
                                                   float* __restrict__ out) {
    const int e  = blockIdx.y * 256 + threadIdx.x;     // column
    const int i0 = blockIdx.x * 1024;                  // 64 row blocks
    float acc = 0.f;
    #pragma unroll 1
    for (int i = i0; i < i0 + 1024; i += 4) {
        acc += attn[i + 0] * anc[(size_t)(i + 0) * EMB + e];
        acc += attn[i + 1] * anc[(size_t)(i + 1) * EMB + e];
        acc += attn[i + 2] * anc[(size_t)(i + 2) * EMB + e];
        acc += attn[i + 3] * anc[(size_t)(i + 3) * EMB + e];
    }
    atomicAdd(&out[e], acc);
}

extern "C" void kernel_launch(void* const* d_in, const int* in_sizes, int n_in,
                              void* d_out, int out_size, void* d_ws, size_t ws_size,
                              hipStream_t stream) {
    const float* leaves    = (const float*)d_in[0];
    const float* ancestors = (const float*)d_in[1];
    const float* W1        = (const float*)d_in[2];
    const float* b1        = (const float*)d_in[3];
    const float* W2        = (const float*)d_in[4];
    const float* b2        = (const float*)d_in[5];
    float* out = (float*)d_out;

    char* ws = (char*)d_ws;
    unsigned short* w1t = (unsigned short*)ws;                          // 2 MB
    float* scores = (float*)(ws + 2u * 1024 * 1024);                    // 256 KB
    float* attn   = (float*)(ws + 2u * 1024 * 1024 + 256u * 1024);      // 256 KB
    float* stats  = (float*)(ws + 2u * 1024 * 1024 + 512u * 1024);      // 8 B

    hipMemsetAsync(d_out, 0, EMB * sizeof(float), stream);

    convert_w1   <<<(K2 * ATT) / 256, 256, 0, stream>>>(W1, w1t);
    scores_kernel<<<NROWS / 128, 256, 0, stream>>>(leaves, ancestors, w1t,
                                                   b1, b2, W2, scores);
    softmax_stats<<<1, 1024, 0, stream>>>(scores, stats);
    attn_kernel  <<<NROWS / 256, 256, 0, stream>>>(scores, stats, attn);
    wsum_kernel  <<<dim3(64, EMB / 256), 256, 0, stream>>>(ancestors, attn, out);
}
